// SwinTransformerModel_13709535609089
// MI455X (gfx1250) — compile-verified, matrix operand order checked by reading, not run on hardware
//
#include <hip/hip_runtime.h>
#include <hip/hip_bf16.h>
#include <math.h>

typedef __attribute__((ext_vector_type(16))) _Float16 v16h;
typedef __attribute__((ext_vector_type(8)))  _Float16 v8h;
typedef __attribute__((ext_vector_type(8)))  float    v8f;
typedef unsigned int u32x4 __attribute__((ext_vector_type(4)));
typedef int          i32x8 __attribute__((ext_vector_type(8)));
typedef int          i32x4 __attribute__((ext_vector_type(4)));

#define HAVE_TDM (__has_builtin(__builtin_amdgcn_tensor_load_to_lds) && \
                  __has_builtin(__builtin_amdgcn_s_wait_tensorcnt))

// ---------------------------------------------------------------------------
// LDS-staged WMMA GEMM: C[M,N] = A[M,K] @ B[K,N] + bias (+ epilogue/residual)
// Block = 256 threads (8 waves), output tile 64x64; wave = 16x32 (2 WMMAs per
// 32-wide K step, A fragment reused). A staged as f16 via coalesced float4
// loads; B tile (32x64 f32) staged by the Tensor Data Mover (tensor_load_to_lds,
// D# per CDNA5 ISA) and converted to f16 at fragment build.
// K mult 32, N mult 64, M mult 64. EPI: 0 = none, 1 = exact GELU.
// ---------------------------------------------------------------------------
template<int EPI>
__global__ __launch_bounds__(256)
void gemm_wmma_kernel(const float* __restrict__ A, const float* __restrict__ B,
                      const float* __restrict__ bias, const float* __restrict__ res,
                      float* __restrict__ C, int M, int N, int K) {
    __shared__ _Float16 Als[64][40];        // [m][k], padded rows (80B pitch)
    __shared__ float    Bf[32][64];         // [k][n], TDM destination
    const int tid  = threadIdx.x;
    const int lane = tid & 31;
    const int wave = tid >> 5;              // 0..7
    const int wm   = wave >> 1;             // 0..3 : 16-row sub-tile
    const int wn   = wave & 1;              // 0..1 : 32-col sub-tile
    const int hf   = lane >> 4;
    const int l16  = lane & 15;
    const int m0   = blockIdx.y * 64;
    const int n0   = blockIdx.x * 64;

    // global staging coordinates (fully coalesced float4 loads)
    const int arow = tid >> 2;              // 0..63
    const int acol = (tid & 3) * 8;         // 0,8,16,24

    v8f acc[2];
#pragma unroll
    for (int ni = 0; ni < 2; ++ni)
#pragma unroll
        for (int r = 0; r < 8; ++r) acc[ni][r] = 0.f;

    for (int k0 = 0; k0 < K; k0 += 32) {
        // ---- stage A tile (64x32) -> f16 LDS -------------------------------
        {
            const float* ap = A + (size_t)(m0 + arow) * K + k0 + acol;
            if (k0 + 32 < K) __builtin_prefetch(ap + 32, 0, 1);
            const float4 f0 = *(const float4*)ap;
            const float4 f1 = *(const float4*)(ap + 4);
            v8h h;
            h[0] = (_Float16)f0.x; h[1] = (_Float16)f0.y;
            h[2] = (_Float16)f0.z; h[3] = (_Float16)f0.w;
            h[4] = (_Float16)f1.x; h[5] = (_Float16)f1.y;
            h[6] = (_Float16)f1.z; h[7] = (_Float16)f1.w;
            *(v8h*)&Als[arow][acol] = h;
        }
        // ---- stage B tile (32x64 f32) via Tensor Data Mover ----------------
#if HAVE_TDM
        if (wave == 0) {
            const unsigned long long gaddr =
                (unsigned long long)(const void*)(B + (size_t)k0 * N + n0);
            const unsigned ldsoff = (unsigned)(size_t)&Bf[0][0];   // aperture low-32 = LDS offset
            u32x4 g0;
            g0[0] = 1u;                                            // count=1, user D#
            g0[1] = ldsoff;                                        // lds_addr
            g0[2] = (unsigned)gaddr;                               // global_addr[31:0]
            g0[3] = (unsigned)((gaddr >> 32) & 0x01FFFFFFu)        // global_addr[56:32]
                  | (2u << 30);                                    // type = 2 ("image")
            i32x8 g1;
            g1[0] = (int)(2u << 16);                               // data_size = 4B
            g1[1] = (int)(((unsigned)N & 0xFFFFu) << 16);          // tensor_dim0 lo16
            g1[2] = (int)(((unsigned)N >> 16) |
                          (((unsigned)K & 0xFFFFu) << 16));        // dim0 hi16 | dim1 lo16
            g1[3] = (int)(((unsigned)K >> 16) | (64u << 16));      // dim1 hi16 | tile_dim0=64
            g1[4] = 32;                                            // tile_dim1=32, tile_dim2=0
            g1[5] = N;                                             // tensor_dim0_stride lo32
            g1[6] = 0;                                             // stride hi | dim1_stride lo
            g1[7] = 0;
            const i32x4 z4 = {0, 0, 0, 0};                         // 2-D tensor: groups 2/3 zero
            const i32x8 z8 = {0, 0, 0, 0, 0, 0, 0, 0};
            __builtin_amdgcn_tensor_load_to_lds(g0, g1, z4, z4, z8, 0);
            __builtin_amdgcn_s_wait_tensorcnt(0);
        }
#else
        {
            const int brow = tid >> 3;                             // 0..31
            const int bcol = (tid & 7) * 8;                        // 0..56
            const float* bp = B + (size_t)(k0 + brow) * N + n0 + bcol;
            *(float4*)&Bf[brow][bcol]     = *(const float4*)bp;
            *(float4*)&Bf[brow][bcol + 4] = *(const float4*)(bp + 4);
        }
#endif
        __syncthreads();

        // ---- fragments ------------------------------------------------------
        const int kb = hf * 8;              // K 0..7 / 8..15 per half-wave
        v16h a;
        {
            const v8h lo = *(const v8h*)&Als[wm * 16 + l16][kb];
            const v8h hi = *(const v8h*)&Als[wm * 16 + l16][kb + 16];
#pragma unroll
            for (int j = 0; j < 8; ++j) { a[j] = lo[j]; a[j + 8] = hi[j]; }
        }
#pragma unroll
        for (int ni = 0; ni < 2; ++ni) {
            const int ncol = wn * 32 + ni * 16 + l16;
            v16h b;
#pragma unroll
            for (int j = 0; j < 8; ++j) {
                b[j]     = (_Float16)Bf[kb + j][ncol];
                b[j + 8] = (_Float16)Bf[kb + 16 + j][ncol];
            }
            acc[ni] = __builtin_amdgcn_wmma_f32_16x16x32_f16(false, a, false, b,
                                                             (short)0, acc[ni], false, false);
        }
        __syncthreads();
    }

    // ---- epilogue ----------------------------------------------------------
#pragma unroll
    for (int ni = 0; ni < 2; ++ni) {
        const int col = n0 + wn * 32 + ni * 16 + l16;
        const float bvl = bias ? bias[col] : 0.f;
#pragma unroll
        for (int r = 0; r < 8; ++r) {
            const int row = m0 + wm * 16 + r + hf * 8;
            float v = acc[ni][r] + bvl;
            if (EPI == 1) v = 0.5f * v * (1.f + erff(v * 0.70710678f));
            if (res) v += res[(size_t)row * N + col];
            C[(size_t)row * N + col] = v;
        }
    }
}

// ---------------------------------------------------------------------------
// Windowed multi-head attention. One block = one window (B*NW = 256 blocks),
// one wave = one head. N=16 tokens, HD=16 -> two WMMAs with K zero-padded
// to 32. Shift/window partition handled by gather/scatter index math.
// ---------------------------------------------------------------------------
__global__ __launch_bounds__(256)
void attn_kernel(const float* __restrict__ qkv, const float* __restrict__ bias_tab,
                 float* __restrict__ out, int shift) {
    __shared__ float S[8][16][16];
    const int wave = threadIdx.x >> 5;      // head 0..7
    const int lane = threadIdx.x & 31;
    const int hf   = lane >> 4;
    const int l16  = lane & 15;
    const int bidx = blockIdx.x;            // 0..255
    const int bb   = bidx >> 4;
    const int w    = bidx & 15;
    const int wy   = w >> 2, wx = w & 3;
    const int hd   = wave;

    auto grow = [&](int tok) -> int {       // window token -> global qkv row
        int y = tok >> 2, x = tok & 3;
        int gy = (wy * 4 + y + shift) & 15;
        int gx = (wx * 4 + x + shift) & 15;
        return bb * 256 + gy * 16 + gx;
    };

    // ---- S = (Q*scale) @ K^T ----------------------------------------------
    v16h aq, bk;
    {
        const int r = grow(l16);
        const float* qp = qkv + (size_t)r * 384 + hd * 16 + hf * 8;
        const float* kp = qkv + (size_t)r * 384 + 128 + hd * 16 + hf * 8;
#pragma unroll
        for (int j = 0; j < 8; ++j) {
            aq[j]     = (_Float16)(qp[j] * 0.25f);   // HD^-0.5 = 0.25
            aq[j + 8] = (_Float16)0.f;               // K pad 16..31
            bk[j]     = (_Float16)kp[j];             // col = key token l16
            bk[j + 8] = (_Float16)0.f;
        }
    }
    v8f s = {0.f,0.f,0.f,0.f,0.f,0.f,0.f,0.f};
    s = __builtin_amdgcn_wmma_f32_16x16x32_f16(false, aq, false, bk,
                                               (short)0, s, false, false);

    // ---- + relative position bias, + shift mask; stage rows in LDS --------
    const int kk = l16, yk = kk >> 2, xk = kk & 3;
#pragma unroll
    for (int r = 0; r < 8; ++r) {
        const int q = r + hf * 8;
        const int yq = q >> 2, xq = q & 3;
        const int rel = (yq - yk + 3) * 7 + (xq - xk + 3);
        float v = s[r] + bias_tab[rel * 8 + hd];
        if (shift) {
            const int hq = wy * 4 + yq, wq = wx * 4 + xq;
            const int hk = wy * 4 + yk, wk = wx * 4 + xk;
            const int idq = (hq < 12 ? 0 : (hq < 14 ? 1 : 2)) * 3 + (wq < 12 ? 0 : (wq < 14 ? 1 : 2));
            const int idk = (hk < 12 ? 0 : (hk < 14 ? 1 : 2)) * 3 + (wk < 12 ? 0 : (wk < 14 ? 1 : 2));
            if (idq != idk) v -= 100.f;
        }
        S[wave][q][kk] = v;
    }
    __syncthreads();

    // ---- softmax per query row; build P fragment ---------------------------
    v16h ap, bv;
    {
        const int q = l16;
        float mx = -1e30f;
#pragma unroll
        for (int j = 0; j < 16; ++j) mx = fmaxf(mx, S[wave][q][j]);
        float e[16], sum = 0.f;
#pragma unroll
        for (int j = 0; j < 16; ++j) { e[j] = __expf(S[wave][q][j] - mx); sum += e[j]; }
        const float inv = 1.f / sum;
#pragma unroll
        for (int j = 0; j < 8; ++j) {
            ap[j]     = (_Float16)(e[hf * 8 + j] * inv);
            ap[j + 8] = (_Float16)0.f;
        }
    }
    // ---- V fragment: col = dim l16, K = key token --------------------------
#pragma unroll
    for (int j = 0; j < 8; ++j) {
        const int key = hf * 8 + j;
        bv[j]     = (_Float16)qkv[(size_t)grow(key) * 384 + 256 + hd * 16 + l16];
        bv[j + 8] = (_Float16)0.f;
    }
    v8f o = {0.f,0.f,0.f,0.f,0.f,0.f,0.f,0.f};
    o = __builtin_amdgcn_wmma_f32_16x16x32_f16(false, ap, false, bv,
                                               (short)0, o, false, false);
#pragma unroll
    for (int r = 0; r < 8; ++r) {
        const int q = r + hf * 8;
        out[(size_t)grow(q) * 128 + hd * 16 + l16] = o[r];
    }
}

// ---------------------------------------------------------------------------
// LayerNorm over C=128. One wave per row, 8 rows per block.
// ---------------------------------------------------------------------------
__global__ __launch_bounds__(256)
void ln_kernel(const float* __restrict__ x, const float* __restrict__ g,
               const float* __restrict__ bta, float* __restrict__ y, int rows) {
    const int wave = threadIdx.x >> 5, lane = threadIdx.x & 31;
    const int row = blockIdx.x * 8 + wave;
    if (row >= rows) return;
    const float* xr = x + (size_t)row * 128;
    float v[4], s = 0.f;
#pragma unroll
    for (int j = 0; j < 4; ++j) { v[j] = xr[lane + 32 * j]; s += v[j]; }
#pragma unroll
    for (int m = 16; m >= 1; m >>= 1) s += __shfl_xor(s, m, 32);
    const float mean = s * (1.f / 128.f);
    float q = 0.f;
#pragma unroll
    for (int j = 0; j < 4; ++j) { const float d = v[j] - mean; q += d * d; }
#pragma unroll
    for (int m = 16; m >= 1; m >>= 1) q += __shfl_xor(q, m, 32);
    const float inv = rsqrtf(q * (1.f / 128.f) + 1e-5f);
    float* yr = y + (size_t)row * 128;
#pragma unroll
    for (int j = 0; j < 4; ++j) {
        const int c = lane + 32 * j;
        yr[c] = (v[j] - mean) * inv * g[c] + bta[c];
    }
}

// ---------------------------------------------------------------------------
// Conv stem: conv(3x3, SAME) + ReLU + maxpool 2x2, fused; one thread/pooled px
// ---------------------------------------------------------------------------
__global__ void conv1_pool_kernel(const float* __restrict__ x, const float* __restrict__ w,
                                  const float* __restrict__ bias, float* __restrict__ y) {
    int idx = blockIdx.x * blockDim.x + threadIdx.x;       // (16,16,128,128)
    if (idx >= 16 * 16 * 128 * 128) return;
    int pw = idx & 127, t = idx >> 7;
    int ph = t & 127; t >>= 7;
    int oc = t & 15;  int b = t >> 4;
    float best = -1e30f;
    for (int dh = 0; dh < 2; ++dh)
    for (int dw = 0; dw < 2; ++dw) {
        const int oh = 2 * ph + dh, ow = 2 * pw + dw;
        float acc = bias[oc];
        for (int ic = 0; ic < 4; ++ic)
            for (int kh = 0; kh < 3; ++kh) {
                const int ih = oh + kh - 1;
                if (ih < 0 || ih >= 256) continue;
                for (int kw = 0; kw < 3; ++kw) {
                    const int iw = ow + kw - 1;
                    if (iw < 0 || iw >= 256) continue;
                    acc += x[(((size_t)b * 4 + ic) * 256 + ih) * 256 + iw] *
                           w[((oc * 4 + ic) * 3 + kh) * 3 + kw];
                }
            }
        best = fmaxf(best, fmaxf(acc, 0.f));
    }
    y[idx] = best;
}

__global__ void conv2_pool_kernel(const float* __restrict__ x, const float* __restrict__ w,
                                  const float* __restrict__ bias, float* __restrict__ y) {
    int idx = blockIdx.x * blockDim.x + threadIdx.x;       // (16,32,64,64)
    if (idx >= 16 * 32 * 64 * 64) return;
    int pw = idx & 63, t = idx >> 6;
    int ph = t & 63; t >>= 6;
    int oc = t & 31; int b = t >> 5;
    float best = -1e30f;
    for (int dh = 0; dh < 2; ++dh)
    for (int dw = 0; dw < 2; ++dw) {
        const int oh = 2 * ph + dh, ow = 2 * pw + dw;
        float acc = bias[oc];
        for (int ic = 0; ic < 16; ++ic)
            for (int kh = 0; kh < 3; ++kh) {
                const int ih = oh + kh - 1;
                if (ih < 0 || ih >= 128) continue;
                for (int kw = 0; kw < 3; ++kw) {
                    const int iw = ow + kw - 1;
                    if (iw < 0 || iw >= 128) continue;
                    acc += x[(((size_t)b * 16 + ic) * 128 + ih) * 128 + iw] *
                           w[((oc * 16 + ic) * 3 + kh) * 3 + kw];
                }
            }
        best = fmaxf(best, fmaxf(acc, 0.f));
    }
    y[idx] = best;
}

// (B,32,64,64) -> (4096, 512); feature f = (py*4+px)*32 + ch
__global__ void patchify_kernel(const float* __restrict__ h, float* __restrict__ A) {
    int idx = blockIdx.x * blockDim.x + threadIdx.x;
    if (idx >= 4096 * 512) return;
    const int f = idx & 511, row = idx >> 9;
    const int ch = f & 31, p = f >> 5, px = p & 3, py = p >> 2;
    const int gx = row & 15, gy = (row >> 4) & 15, b = row >> 8;
    A[idx] = h[(((size_t)b * 32 + ch) * 64 + (gy * 4 + py)) * 64 + (gx * 4 + px)];
}

// 128 -> 2 head, per token
__global__ void head_kernel(const float* __restrict__ u, const float* __restrict__ w,
                            const float* __restrict__ bias, float* __restrict__ logits) {
    int idx = blockIdx.x * blockDim.x + threadIdx.x;       // 4096*2
    if (idx >= 8192) return;
    const int o = idx & 1, row = idx >> 1;
    float acc = bias[o];
    for (int c = 0; c < 128; ++c) acc += u[(size_t)row * 128 + c] * w[c * 2 + o];
    logits[idx] = acc;
}

// nearest x16 upsample + sigmoid -> (16,2,256,256)
__global__ void upsample_kernel(const float* __restrict__ logits, float* __restrict__ out) {
    int idx = blockIdx.x * blockDim.x + threadIdx.x;
    if (idx >= 16 * 2 * 256 * 256) return;
    const int wc = idx & 255; int t = idx >> 8;
    const int hr = t & 255; t >>= 8;
    const int o = t & 1, b = t >> 1;
    const int tok = (hr >> 4) * 16 + (wc >> 4);
    const float v = logits[((size_t)b * 256 + tok) * 2 + o];
    out[idx] = 1.f / (1.f + __expf(-v));
}

// ---------------------------------------------------------------------------
extern "C" void kernel_launch(void* const* d_in, const int* in_sizes, int n_in,
                              void* d_out, int out_size, void* d_ws, size_t ws_size,
                              hipStream_t stream) {
    (void)in_sizes; (void)n_in; (void)out_size;
    const float* x       = (const float*)d_in[0];
    const float* conv1_w = (const float*)d_in[1];
    const float* conv1_b = (const float*)d_in[2];
    const float* conv2_w = (const float*)d_in[3];
    const float* conv2_b = (const float*)d_in[4];
    const float* pe_w    = (const float*)d_in[5];
    const float* pe_b    = (const float*)d_in[6];
    const float* peln_g  = (const float*)d_in[7];
    const float* peln_b  = (const float*)d_in[8];
    const float* ln1_g   = (const float*)d_in[9];
    const float* ln1_b   = (const float*)d_in[10];
    const float* qkv_w   = (const float*)d_in[11];
    const float* qkv_b   = (const float*)d_in[12];
    const float* proj_w  = (const float*)d_in[13];
    const float* proj_b  = (const float*)d_in[14];
    const float* btab    = (const float*)d_in[15];
    const float* ln2_g   = (const float*)d_in[16];
    const float* ln2_b   = (const float*)d_in[17];
    const float* mlp_w1  = (const float*)d_in[18];
    const float* mlp_b1  = (const float*)d_in[19];
    const float* mlp_w2  = (const float*)d_in[20];
    const float* mlp_b2  = (const float*)d_in[21];
    const float* fn_g    = (const float*)d_in[22];
    const float* fn_b    = (const float*)d_in[23];
    const float* head_w  = (const float*)d_in[24];
    const float* head_b  = (const float*)d_in[25];
    float* out = (float*)d_out;

    // workspace layout (float offsets)
    float* ws = (float*)d_ws;
    const size_t o_conv1 = 0;                      // 16*16*128*128 = 4,194,304
    const size_t o_conv2 = o_conv1 + 4194304;      // 16*32*64*64   = 2,097,152
    const size_t o_A     = o_conv2 + 2097152;      // 4096*512      = 2,097,152
    const size_t o_t     = o_A     + 2097152;      // 4096*128
    const size_t o_u     = o_t     + 524288;
    const size_t o_qkv   = o_u     + 524288;       // 4096*384
    const size_t o_attn  = o_qkv   + 1572864;      // 4096*128
    const size_t o_mlp   = o_attn  + 524288;       // 4096*512
    const size_t o_logit = o_mlp   + 2097152;      // 4096*2
    const size_t need    = (o_logit + 8192) * sizeof(float);
    if (ws_size < need) return;

    float* c1  = ws + o_conv1;
    float* c2  = ws + o_conv2;
    float* Ap  = ws + o_A;
    float* t   = ws + o_t;
    float* u   = ws + o_u;
    float* qkv = ws + o_qkv;
    float* ao  = ws + o_attn;
    float* mh  = ws + o_mlp;
    float* lg  = ws + o_logit;

    const int M = 4096;
    // stem
    conv1_pool_kernel<<<(4194304 + 255) / 256, 256, 0, stream>>>(x, conv1_w, conv1_b, c1);
    conv2_pool_kernel<<<(2097152 + 255) / 256, 256, 0, stream>>>(c1, conv2_w, conv2_b, c2);
    patchify_kernel<<<(2097152 + 255) / 256, 256, 0, stream>>>(c2, Ap);
    // patch embed + LN  (grid: N/64 x M/64, 256 threads)
    gemm_wmma_kernel<0><<<dim3(128 / 64, M / 64), 256, 0, stream>>>(Ap, pe_w, pe_b, nullptr, u, M, 128, 512);
    ln_kernel<<<M / 8, 256, 0, stream>>>(u, peln_g, peln_b, t, M);

    for (int i = 0; i < 12; ++i) {
        const int shift = (i & 1) ? 2 : 0;
        ln_kernel<<<M / 8, 256, 0, stream>>>(t, ln1_g + i * 128, ln1_b + i * 128, u, M);
        gemm_wmma_kernel<0><<<dim3(384 / 64, M / 64), 256, 0, stream>>>(
            u, qkv_w + (size_t)i * 128 * 384, qkv_b + i * 384, nullptr, qkv, M, 384, 128);
        attn_kernel<<<256, 256, 0, stream>>>(qkv, btab + (size_t)i * 49 * 8, ao, shift);
        gemm_wmma_kernel<0><<<dim3(128 / 64, M / 64), 256, 0, stream>>>(
            ao, proj_w + (size_t)i * 128 * 128, proj_b + i * 128, t, t, M, 128, 128);
        ln_kernel<<<M / 8, 256, 0, stream>>>(t, ln2_g + i * 128, ln2_b + i * 128, u, M);
        gemm_wmma_kernel<1><<<dim3(512 / 64, M / 64), 256, 0, stream>>>(
            u, mlp_w1 + (size_t)i * 128 * 512, mlp_b1 + i * 512, nullptr, mh, M, 512, 128);
        gemm_wmma_kernel<0><<<dim3(128 / 64, M / 64), 256, 0, stream>>>(
            mh, mlp_w2 + (size_t)i * 512 * 128, mlp_b2 + i * 128, t, t, M, 128, 512);
    }

    ln_kernel<<<M / 8, 256, 0, stream>>>(t, fn_g, fn_b, u, M);
    head_kernel<<<8192 / 256, 256, 0, stream>>>(u, head_w, head_b, lg);
    upsample_kernel<<<(2097152 + 255) / 256, 256, 0, stream>>>(lg, out);
}
